// KernelAgg_49744311222729
// MI455X (gfx1250) — matrix operand order repaired and checked by reading, not compiled, not run on hardware
//
#include <hip/hip_runtime.h>
#include <hip/hip_bf16.h>
#include <stdint.h>

// Shapes fixed by the reference: B=256, N=512, T=1, dx=32, D=512.
#define B_    256
#define N_    512
#define DX_   32
#define D_    512
#define LAMBDA_REG 0.1f
#define CG_ITERS   32
#define NWAVES    (N_ / 32)     // 16 waves of 32 lanes per CG workgroup

typedef __attribute__((ext_vector_type(16))) __bf16 v16bf;
typedef __attribute__((ext_vector_type(8)))  float  v8f;

struct Pack32 { uint4 a, b; };   // 32 bytes == v16bf

// ---------------------------------------------------------------------------
// Phase 0: bf16 copy of context rows, f32 row norms, f32 rhs = k_star(:,0)
// ---------------------------------------------------------------------------
__global__ __launch_bounds__(256) void prep_kernel(
    const float* __restrict__ ctx,      // [B, N, DX]
    const float* __restrict__ tgt,      // [B, 1, DX]
    const float* __restrict__ lsp,      // [1]
    __hip_bfloat16* __restrict__ xb,    // [B, N, DX] bf16
    float* __restrict__ norms,          // [B, N]
    float* __restrict__ rhs)            // [B, N]
{
  int idx = blockIdx.x * blockDim.x + threadIdx.x;   // b*N + n
  if (idx >= B_ * N_) return;
  int b = idx / N_;
  float ls   = lsp[0];
  float inv2 = -0.5f / (ls * ls);
  const float* xr = ctx + (size_t)idx * DX_;
  const float* xt = tgt + (size_t)b   * DX_;
  float nrm = 0.f, dist = 0.f;
  #pragma unroll
  for (int k = 0; k < DX_; ++k) {
    float v = xr[k];
    nrm += v * v;
    float dd = v - xt[k];
    dist += dd * dd;
    xb[(size_t)idx * DX_ + k] = __float2bfloat16(v);
  }
  norms[idx] = nrm;
  rhs[idx]   = expf(dist * inv2);
}

// ---------------------------------------------------------------------------
// Phase 1: K_reg = exp(-0.5*(n_i + n_j - 2*X X^T)/ls^2) + lambda*I  (bf16)
// One wave32 computes one 16x16 tile with a single v_wmma_f32_16x16x32_bf16
// (dx == 32 == WMMA K-dim). 1024 tiles per batch, 8 waves per block.
// ---------------------------------------------------------------------------
__global__ __launch_bounds__(256) void build_k_kernel(
    const __hip_bfloat16* __restrict__ xb,   // [B, N, 32] bf16
    const float* __restrict__ norms,         // [B, N]
    const float* __restrict__ lsp,           // [1]
    __hip_bfloat16* __restrict__ K)          // [B, N, N] bf16
{
  int b    = blockIdx.y;
  int wave = threadIdx.x >> 5;
  int lane = threadIdx.x & 31;
  int tile = blockIdx.x * 8 + wave;      // 0..1023
  int ti = tile >> 5, tj = tile & 31;
  int i0 = ti * 16,  j0 = tj * 16;
  int sub = lane & 15;                   // M (or N) index within tile
  int hi  = lane >> 4;                   // lane-half selector

  // A operand: rows i0..i0+15 of X. Per ISA 16-bit A 16x32 layout:
  // lane half 0 -> K {0..7, 16..23}, half 1 -> K {8..15, 24..31}.
  const uint4* pa = (const uint4*)(xb + ((size_t)b * N_ + (i0 + sub)) * DX_);
  Pack32 apk; apk.a = pa[hi]; apk.b = pa[hi + 2];
  v16bf av = __builtin_bit_cast(v16bf, apk);

  // B operand: B[k][n] = X[j0+n][k], same K order as A:
  // lane half 0 -> K {0..7, 16..23}, half 1 -> K {8..15, 24..31} of column n = lane&15.
  const uint4* pb = (const uint4*)(xb + ((size_t)b * N_ + (j0 + sub)) * DX_);
  Pack32 bpk; bpk.a = pb[hi]; bpk.b = pb[hi + 2];
  v16bf bv = __builtin_bit_cast(v16bf, bpk);

  v8f c = {};
  c = __builtin_amdgcn_wmma_f32_16x16x32_bf16(
        /*neg_a=*/false, av, /*neg_b=*/false, bv,
        /*c_mod=*/(short)0, c, /*reuse_a=*/false, /*reuse_b=*/false);

  float ls   = lsp[0];
  float inv2 = -0.5f / (ls * ls);
  int   gj   = j0 + sub;
  float nj   = norms[(size_t)b * N_ + gj];
  __hip_bfloat16* kb = K + (size_t)b * N_ * N_;
  // C layout: VGPR r, lanes 0-15 -> M=r, lanes 16-31 -> M=r+8; col = lane&15.
  #pragma unroll
  for (int r = 0; r < 8; ++r) {
    int   gi  = i0 + r + hi * 8;
    float ni  = norms[(size_t)b * N_ + gi];
    float sq  = fmaxf(ni + nj - 2.f * c[r], 0.f);
    float val = expf(sq * inv2);
    if (gi == gj) val += LAMBDA_REG;           // + lambda * I
    kb[(size_t)gi * N_ + gj] = __float2bfloat16(val);
  }
}

// ---------------------------------------------------------------------------
// Phase 2: per-batch CG solve (SPD, diag~1.1 -> fast convergence),
// then softmax over N and out[b,d] = sum_n w_n * enc[b,n,d].
// One 512-thread workgroup (16 wave32s) per batch; K rows stream from the
// 192MB L2 (whole bf16 K is 128MB and stays resident across all 32 sweeps).
// ---------------------------------------------------------------------------
__device__ __forceinline__ float wave_reduce_sum(float v) {
  #pragma unroll
  for (int s = 16; s > 0; s >>= 1) v += __shfl_xor(v, s, 32);
  return v;
}
__device__ __forceinline__ float wave_reduce_max(float v) {
  #pragma unroll
  for (int s = 16; s > 0; s >>= 1) v = fmaxf(v, __shfl_xor(v, s, 32));
  return v;
}

// Block reductions: wave shuffle + NWAVES partials in LDS + broadcast read.
__device__ __forceinline__ float breduce_sum(float v, float* buf) {
  int t = threadIdx.x;
  float w = wave_reduce_sum(v);
  if ((t & 31) == 0) buf[t >> 5] = w;
  __syncthreads();
  float s = 0.f;
  #pragma unroll
  for (int i = 0; i < NWAVES; ++i) s += buf[i];   // broadcast reads
  __syncthreads();
  return s;
}
__device__ __forceinline__ float breduce_max(float v, float* buf) {
  int t = threadIdx.x;
  float w = wave_reduce_max(v);
  if ((t & 31) == 0) buf[t >> 5] = w;
  __syncthreads();
  float s = -__builtin_inff();
  #pragma unroll
  for (int i = 0; i < NWAVES; ++i) s = fmaxf(s, buf[i]);
  __syncthreads();
  return s;
}

__device__ __forceinline__ float bf16_lo(uint32_t u) { return __uint_as_float(u << 16); }
__device__ __forceinline__ float bf16_hi(uint32_t u) { return __uint_as_float(u & 0xffff0000u); }

__global__ __launch_bounds__(512) void cg_softmax_kernel(
    const __hip_bfloat16* __restrict__ K,    // [B, N, N] bf16
    const float* __restrict__ rhs,           // [B, N]
    const float* __restrict__ enc,           // [B, N, D]
    float* __restrict__ out)                 // [B, D]
{
  __shared__ float p_s[N_];
  __shared__ float Ap_s[N_];
  __shared__ float wbuf[NWAVES];
  int b    = blockIdx.x;
  int n    = threadIdx.x;
  int lane = n & 31;
  int wid  = n >> 5;

  const __hip_bfloat16* Kb = K + (size_t)b * N_ * N_;

  float r  = rhs[(size_t)b * N_ + n];
  float x  = 0.f;
  p_s[n]   = r;                       // covered by breduce's first barrier
  float rz = breduce_sum(r * r, wbuf);

  for (int it = 0; it < CG_ITERS; ++it) {
    float p = p_s[n];

    // --- Ap = K p, row-cooperative: wave `wid` owns rows wid*32..wid*32+31.
    // Each row is 64 uint4; lane l takes chunks l and l+32 (two fully
    // coalesced 512B wave loads), then a 5-step shuffle reduction.
    // p chunk per lane is row-invariant -> hoisted out of the row loop.
    float pc[16];
    #pragma unroll
    for (int k = 0; k < 8; ++k) {
      pc[k]     = p_s[lane * 8 + k];
      pc[8 + k] = p_s[256 + lane * 8 + k];
    }
    int row0 = wid * 32;
    #pragma unroll 4
    for (int rr = 0; rr < 32; ++rr) {
      int i = row0 + rr;
      const uint4* kr = (const uint4*)(Kb + (size_t)i * N_);
      uint4 u0 = kr[lane];
      uint4 u1 = kr[lane + 32];
      float s = 0.f;
      s = fmaf(bf16_lo(u0.x), pc[0],  s);
      s = fmaf(bf16_hi(u0.x), pc[1],  s);
      s = fmaf(bf16_lo(u0.y), pc[2],  s);
      s = fmaf(bf16_hi(u0.y), pc[3],  s);
      s = fmaf(bf16_lo(u0.z), pc[4],  s);
      s = fmaf(bf16_hi(u0.z), pc[5],  s);
      s = fmaf(bf16_lo(u0.w), pc[6],  s);
      s = fmaf(bf16_hi(u0.w), pc[7],  s);
      s = fmaf(bf16_lo(u1.x), pc[8],  s);
      s = fmaf(bf16_hi(u1.x), pc[9],  s);
      s = fmaf(bf16_lo(u1.y), pc[10], s);
      s = fmaf(bf16_hi(u1.y), pc[11], s);
      s = fmaf(bf16_lo(u1.z), pc[12], s);
      s = fmaf(bf16_hi(u1.z), pc[13], s);
      s = fmaf(bf16_lo(u1.w), pc[14], s);
      s = fmaf(bf16_hi(u1.w), pc[15], s);
      s = wave_reduce_sum(s);
      if (lane == 0) Ap_s[i] = s;
    }
    __syncthreads();
    float Ap = Ap_s[n];

    float pAp   = breduce_sum(p * Ap, wbuf);
    float alpha = rz / pAp;
    x += alpha * p;
    r -= alpha * Ap;
    float rz_new = breduce_sum(r * r, wbuf);
    float beta   = rz_new / rz;
    rz = rz_new;
    p_s[n] = r + beta * p;            // all p_s reads done (breduce barriers)
    __syncthreads();
  }

  // softmax over context dim
  float m = breduce_max(x, wbuf);
  float e = expf(x - m);
  float s = breduce_sum(e, wbuf);
  p_s[n]  = e / s;                    // reuse LDS for softmax weights
  __syncthreads();

  // out[b, d] = sum_n w[n] * enc[b, n, d]   (d == threadIdx.x, coalesced)
  const float* encb = enc + (size_t)b * N_ * D_;
  float acc = 0.f;
  #pragma unroll 4
  for (int j = 0; j < N_; ++j)
    acc = fmaf(p_s[j], encb[(size_t)j * D_ + n], acc);
  out[(size_t)b * D_ + n] = acc;
}

// ---------------------------------------------------------------------------
// Workspace layout (requires ws_size >= ~137 MB):
//   [0,   8MB)   Xbf16  [B,N,32]
//   [8MB, 8.5MB) norms [B,N] f32
//   [8.5MB, 9MB) rhs   [B,N] f32
//   [9MB, 137MB) K     [B,N,N] bf16  (fits in MI455X 192MB L2 during CG)
// ---------------------------------------------------------------------------
extern "C" void kernel_launch(void* const* d_in, const int* in_sizes, int n_in,
                              void* d_out, int out_size, void* d_ws, size_t ws_size,
                              hipStream_t stream) {
  const float* ctx = (const float*)d_in[0];   // context_xi [B,N,dx]
  const float* tgt = (const float*)d_in[1];   // target_xi  [B,1,dx]
  const float* enc = (const float*)d_in[2];   // encoded    [B,N,D]
  const float* ls  = (const float*)d_in[3];   // lengthscale[1]
  float* out = (float*)d_out;                 // [B,D] f32

  char* ws = (char*)d_ws;
  __hip_bfloat16* xb    = (__hip_bfloat16*)(ws);
  float*          norms = (float*)(ws + (size_t)8 * 1024 * 1024);
  float*          rhs   = (float*)(ws + (size_t)8 * 1024 * 1024 + 512 * 1024);
  __hip_bfloat16* K     = (__hip_bfloat16*)(ws + (size_t)9 * 1024 * 1024);

  prep_kernel<<<(B_ * N_) / 256, 256, 0, stream>>>(ctx, tgt, ls, xb, norms, rhs);
  build_k_kernel<<<dim3(128, B_), 256, 0, stream>>>(xb, norms, ls, K);
  cg_softmax_kernel<<<B_, N_, 0, stream>>>(K, rhs, enc, out);
}
